// STransformer_45217415692338
// MI455X (gfx1250) — matrix operand order repaired and checked by reading, not compiled, not run on hardware
//
#include <hip/hip_runtime.h>
#include <hip/hip_bf16.h>

// ---------------------------------------------------------------------------
// STransformer forward for MI455X (gfx1250, wave32, WMMA).
// All GEMM work runs through v_wmma_f32_16x16x32_bf16 (bf16 in, f32 acc).
// Attention / bf16-GEMM tile staging uses GLOBAL_LOAD_ASYNC_TO_LDS_B128
// (ASYNCcnt) when the toolchain exposes the builtins.
// ---------------------------------------------------------------------------

typedef __attribute__((ext_vector_type(16))) __bf16 v16bf;
typedef __attribute__((ext_vector_type(8)))  float  v8f;
typedef __attribute__((ext_vector_type(4)))  int    v4i;

#define DEVINL static __device__ __forceinline__

constexpr int  Bb = 8, Nn = 128, Tt = 64, Cc = 256, Hh = 8, FEe = 4;
constexpr long long SZ = (long long)Bb * Nn * Tt * Cc;   // 16,777,216 elements

#if defined(__HIP_DEVICE_COMPILE__) && defined(__gfx1250__) && \
    __has_builtin(__builtin_amdgcn_global_load_async_to_lds_b128) && \
    __has_builtin(__builtin_amdgcn_s_wait_asynccnt)
#define USE_ASYNC_LDS 1
#else
#define USE_ASYNC_LDS 0
#endif

#if USE_ASYNC_LDS
// builtin signature (from hipcc diagnostic): (v4i AS1*, v4i AS3*, Imm, Imm)
#define ASYNC_B128(gptr, lptr)                                                \
  __builtin_amdgcn_global_load_async_to_lds_b128(                             \
      (__attribute__((address_space(1))) v4i*)(v4i*)(void*)(gptr),            \
      (__attribute__((address_space(3))) v4i*)(v4i*)(void*)(lptr), 0, 0)
#endif

DEVINL v8f wmma_bf16(v16bf a, v16bf b, v8f c) {
  return __builtin_amdgcn_wmma_f32_16x16x32_bf16(false, a, false, b,
                                                 (short)0, c, false, false);
}
DEVINL __bf16 f2bf(float x) { return (__bf16)x; }
DEVINL unsigned pack2(float lo, float hi) {        // -> v_cvt_pk_bf16_f32
  unsigned short a = __builtin_bit_cast(unsigned short, (__bf16)lo);
  unsigned short b = __builtin_bit_cast(unsigned short, (__bf16)hi);
  return (unsigned)a | ((unsigned)b << 16);
}

DEVINL float rmax16(float x) {
#pragma unroll
  for (int m = 1; m < 16; m <<= 1) x = fmaxf(x, __shfl_xor(x, m, 32));
  return x;
}
DEVINL float rsum16(float x) {
#pragma unroll
  for (int m = 1; m < 16; m <<= 1) x += __shfl_xor(x, m, 32);
  return x;
}
DEVINL float rsum32(float x) {
#pragma unroll
  for (int m = 1; m < 32; m <<= 1) x += __shfl_xor(x, m, 32);
  return x;
}

// ---------------------------------------------------------------------------
// Batched tiled GEMM: C[b] = A[b] * B[b] (+bias, +relu), bf16 WMMA.
// Block tile 128x128, K-step 32, double-buffered LDS with register staging.
// 8 waves: wave (wid&3, wid>>2) owns a 32x64 sub-tile -> 8 WMMAs / K-step.
// EPI: 0 = none, 1 = +bias[col & biasMask], 2 = relu(+bias).
// ---------------------------------------------------------------------------
template <int EPI, bool A_BF16, bool OUT_BF16>
__global__ __launch_bounds__(256) void gemm_wmma_k(
    const void* __restrict__ Av, const float* __restrict__ B,
    void* __restrict__ Cv, const float* __restrict__ bias,
    int M, int N, int K, int lda, int ldb, int ldc,
    long long sA, long long sB, long long sC, int biasMask) {
  __shared__ __bf16 As[2][128][40];     // 20.0 KB (pitch 80B, 16B-aligned)
  __shared__ __bf16 Bs[2][32][136];     // 17.0 KB (pitch 272B, 16B-aligned)

  const int tid = threadIdx.x, lane = tid & 31, wid = tid >> 5;
  const int wm = (wid & 3) * 32;
  const int wn = (wid >> 2) * 64;
  const long long tileM = (long long)blockIdx.y * 128;
  const long long tileN = (long long)blockIdx.x * 128;
  const long long zb = blockIdx.z;

  const float*  Af = (const float*)Av;
  const __bf16* Ab = (const __bf16*)Av;
  const long long abase = sA * zb + tileM * lda;
  const long long bbase = sB * zb + tileN;

  float    ra[16];
  unsigned rau[8];
  float    rb[16];

  auto loadA = [&](int k0) {
    if constexpr (A_BF16) {
#if USE_ASYNC_LDS
      (void)rau;  // async path DMAs straight into LDS at storeLds() time
#else
#pragma unroll
      for (int j = 0; j < 8; ++j) {
        const int r = (tid >> 4) + j * 16, c2 = tid & 15;
        rau[j] = *(const unsigned*)&Ab[abase + (long long)r * lda + k0 + 2 * c2];
      }
#endif
    } else {
#pragma unroll
      for (int j = 0; j < 8; ++j) {
        const int r = (tid >> 4) + j * 16, c2 = tid & 15;
        const float* p = &Af[abase + (long long)r * lda + k0 + 2 * c2];
        ra[2 * j] = p[0];
        ra[2 * j + 1] = p[1];
      }
    }
  };
  auto loadB = [&](int k0) {
#pragma unroll
    for (int j = 0; j < 8; ++j) {
      const int r = (tid >> 6) + j * 4, c2 = tid & 63;
      const float* p = &B[bbase + (long long)(k0 + r) * ldb + 2 * c2];
      rb[2 * j] = p[0];
      rb[2 * j + 1] = p[1];
    }
  };
  auto storeLds = [&](int buf, int k0) {
    if constexpr (A_BF16) {
#if USE_ASYNC_LDS
      // bf16 A tile: pure byte copy -> async DMA into LDS (16B per lane/chunk)
#pragma unroll
      for (int j = 0; j < 2; ++j) {                    // 512 chunks of 16B
        const int c = tid + j * 256;
        const int r = c >> 2, q8 = (c & 3) * 8;
        ASYNC_B128(&Ab[abase + (long long)r * lda + k0 + q8], &As[buf][r][q8]);
      }
#else
#pragma unroll
      for (int j = 0; j < 8; ++j) {
        const int r = (tid >> 4) + j * 16, c2 = tid & 15;
        *(unsigned*)&As[buf][r][2 * c2] = rau[j];
      }
#endif
    } else {
#pragma unroll
      for (int j = 0; j < 8; ++j) {
        const int r = (tid >> 4) + j * 16, c2 = tid & 15;
        *(unsigned*)&As[buf][r][2 * c2] = pack2(ra[2 * j], ra[2 * j + 1]);
      }
    }
#pragma unroll
    for (int j = 0; j < 8; ++j) {
      const int r = (tid >> 6) + j * 4, c2 = tid & 63;
      *(unsigned*)&Bs[buf][r][2 * c2] = pack2(rb[2 * j], rb[2 * j + 1]);
    }
  };

  const v8f vz = {0.f, 0.f, 0.f, 0.f, 0.f, 0.f, 0.f, 0.f};
  v8f acc[2][4];
#pragma unroll
  for (int i = 0; i < 2; ++i)
#pragma unroll
    for (int j = 0; j < 4; ++j) acc[i][j] = vz;

  const int KT = K >> 5;
  loadA(0);
  loadB(0);
  storeLds(0, 0);
#if USE_ASYNC_LDS
  if (A_BF16) __builtin_amdgcn_s_wait_asynccnt(0);
#endif
  __syncthreads();

  for (int kt = 0; kt < KT; ++kt) {
    const int cur = kt & 1;
    if (kt + 1 < KT) {        // issue next-step global loads before the math
      loadA((kt + 1) << 5);
      loadB((kt + 1) << 5);
    }

    v16bf a0, a1;
#pragma unroll
    for (int e = 0; e < 16; ++e) {
      const int kk = (e < 8 ? e : e + 8) + 8 * (lane >> 4);
      a0[e] = As[cur][wm + (lane & 15)][kk];
      a1[e] = As[cur][wm + 16 + (lane & 15)][kk];
    }
#pragma unroll
    for (int j = 0; j < 4; ++j) {
      v16bf bb;
#pragma unroll
      for (int e = 0; e < 16; ++e) bb[e] = Bs[cur][(e < 8 ? e : e + 8) + 8 * (lane >> 4)][wn + j * 16 + (lane & 15)];
      acc[0][j] = wmma_bf16(a0, bb, acc[0][j]);
      acc[1][j] = wmma_bf16(a1, bb, acc[1][j]);
    }

    if (kt + 1 < KT) storeLds(cur ^ 1, (kt + 1) << 5);
#if USE_ASYNC_LDS
    if (A_BF16) __builtin_amdgcn_s_wait_asynccnt(0);
#endif
    __syncthreads();
  }

#pragma unroll
  for (int i = 0; i < 2; ++i)
#pragma unroll
    for (int j = 0; j < 4; ++j)
#pragma unroll
      for (int r = 0; r < 8; ++r) {
        const long long row = tileM + wm + i * 16 + r + 8 * (lane >> 4);
        const int col = (int)tileN + wn + j * 16 + (lane & 15);
        float v = acc[i][j][r];
        if (EPI >= 1) v += bias[col & biasMask];
        if (EPI == 2) v = fmaxf(v, 0.f);
        const long long idx = sC * zb + row * ldc + col;
        if (OUT_BF16) ((__bf16*)Cv)[idx] = f2bf(v);
        else          ((float*)Cv)[idx]  = v;
      }
}

// ---------------------------------------------------------------------------
// Per-head projection: Y[row, h*32+e] = sum_d X[row, h*32+d] * W[d,e].
// Block = 64 rows x all 8 heads; wave w == head w. K = 32 -> 1 WMMA step.
// ---------------------------------------------------------------------------
__global__ __launch_bounds__(256) void proj_heads_k(
    const float* __restrict__ X, const float* __restrict__ W,
    __bf16* __restrict__ Y) {
  __shared__ __bf16 Xs[64][Cc + 8];
  __shared__ __bf16 Ws[32][36];
  const int tid = threadIdx.x, lane = tid & 31, h = tid >> 5;
  const long long row0 = (long long)blockIdx.x * 64;

#pragma unroll
  for (int i = tid; i < 32 * 32; i += 256) Ws[i >> 5][i & 31] = f2bf(W[i]);
#pragma unroll
  for (int i = tid; i < 64 * Cc / 2; i += 256) {   // pair-packed staging
    const int r = i >> 7, c2 = i & 127;
    const float* p = &X[(row0 + r) * Cc + 2 * c2];
    *(unsigned*)&Xs[r][2 * c2] = pack2(p[0], p[1]);
  }
  __syncthreads();

  v16bf b0, b1;
#pragma unroll
  for (int e = 0; e < 16; ++e) { const int kk = (e < 8 ? e : e + 8) + 8 * (lane >> 4); b0[e] = Ws[kk][lane & 15]; b1[e] = Ws[kk][16 + (lane & 15)]; }

#pragma unroll
  for (int rg = 0; rg < 4; ++rg) {
    v16bf a;
#pragma unroll
    for (int e = 0; e < 16; ++e) {
      const int kk = (e < 8 ? e : e + 8) + 8 * (lane >> 4);
      a[e] = Xs[rg * 16 + (lane & 15)][h * 32 + kk];
    }
    const v8f vz = {0.f, 0.f, 0.f, 0.f, 0.f, 0.f, 0.f, 0.f};
    v8f c0 = wmma_bf16(a, b0, vz);
    v8f c1 = wmma_bf16(a, b1, vz);
#pragma unroll
    for (int r = 0; r < 8; ++r) {
      const long long row = row0 + rg * 16 + r + 8 * (lane >> 4);
      Y[row * Cc + h * 32 + (lane & 15)]      = f2bf(c0[r]);
      Y[row * Cc + h * 32 + 16 + (lane & 15)] = f2bf(c1[r]);
    }
  }
}

// ---------------------------------------------------------------------------
// Spatial attention per (b,t,h): S = Q K^T / sqrt(C); softmax over key nodes;
// O = P V.  One workgroup per (h,t,b); wave w owns S rows [16w, 16w+16).
// Q/K/V slabs (128x32 bf16) staged via async LDS DMA when available.
// ---------------------------------------------------------------------------
__global__ __launch_bounds__(256) void attn_k(
    const __bf16* __restrict__ Qp, const __bf16* __restrict__ Kp,
    const __bf16* __restrict__ Vp, float* __restrict__ O) {
  __shared__ __bf16 Qs[Nn][40], Ks[Nn][40], Vs[Nn][40];   // 16B-aligned rows
  __shared__ __bf16 Pb[Nn][136];

  const int h = blockIdx.x, t = blockIdx.y, b = blockIdx.z;
  const int tid = threadIdx.x, lane = tid & 31, wid = tid >> 5;

#if USE_ASYNC_LDS
#pragma unroll
  for (int j = 0; j < 2; ++j) {               // 512 chunks of 16B per matrix
    const int c = tid + j * 256;
    const int n = c >> 2, q8 = (c & 3) * 8;
    const long long g = (((long long)(b * Nn + n)) * Tt + t) * Cc + h * 32 + q8;
    ASYNC_B128(Qp + g, &Qs[n][q8]);
    ASYNC_B128(Kp + g, &Ks[n][q8]);
    ASYNC_B128(Vp + g, &Vs[n][q8]);
  }
  __builtin_amdgcn_s_wait_asynccnt(0);
#else
#pragma unroll
  for (int i = tid; i < Nn * 32; i += 256) {
    const int n = i >> 5, cc = i & 31;
    const long long g = (((long long)(b * Nn + n)) * Tt + t) * Cc + h * 32 + cc;
    Qs[n][cc] = Qp[g];
    Ks[n][cc] = Kp[g];
    Vs[n][cc] = Vp[g];
  }
#endif
  __syncthreads();

  const v8f vz = {0.f, 0.f, 0.f, 0.f, 0.f, 0.f, 0.f, 0.f};
  v8f s[8];
#pragma unroll
  for (int i = 0; i < 8; ++i) s[i] = vz;

  v16bf a;
#pragma unroll
  for (int e = 0; e < 16; ++e) {
    const int kk = (e < 8 ? e : e + 8) + 8 * (lane >> 4);
    a[e] = Qs[wid * 16 + (lane & 15)][kk];
  }
#pragma unroll
  for (int nt = 0; nt < 8; ++nt) {            // S = Q * K^T (B frag: n = lane&15, k table order)
    v16bf bb;
#pragma unroll
    for (int e = 0; e < 16; ++e) bb[e] = Ks[nt * 16 + (lane & 15)][(e < 8 ? e : e + 8) + 8 * (lane >> 4)];
    s[nt] = wmma_bf16(a, bb, s[nt]);
  }

  const float scale = 0.0625f;                // 1/sqrt(C)
#pragma unroll
  for (int r = 0; r < 8; ++r) {               // row softmax over 128 key nodes
    float mx = -3.0e38f;
#pragma unroll
    for (int nt = 0; nt < 8; ++nt) mx = fmaxf(mx, s[nt][r]);
    mx = rmax16(mx) * scale;
    float sm = 0.f;
#pragma unroll
    for (int nt = 0; nt < 8; ++nt) {
      const float e_ = __expf(s[nt][r] * scale - mx);
      s[nt][r] = e_;
      sm += e_;
    }
    sm = rsum16(sm);
    const float inv = 1.f / sm;
#pragma unroll
    for (int nt = 0; nt < 8; ++nt) s[nt][r] *= inv;
  }
#pragma unroll
  for (int nt = 0; nt < 8; ++nt)
#pragma unroll
    for (int r = 0; r < 8; ++r)
      Pb[wid * 16 + r + 8 * (lane >> 4)][nt * 16 + (lane & 15)] = f2bf(s[nt][r]);
  __syncthreads();

  v8f o0 = vz, o1 = vz;                       // O = P * V (K = 128, 4 steps)
#pragma unroll
  for (int kk = 0; kk < 4; ++kk) {
    v16bf ap, b0, b1;
#pragma unroll
    for (int e = 0; e < 16; ++e) {
      const int ko = (e < 8 ? e : e + 8) + 8 * (lane >> 4);
      ap[e] = Pb[wid * 16 + (lane & 15)][kk * 32 + ko];
    }
#pragma unroll
    for (int e = 0; e < 16; ++e) {
      const int ko = (e < 8 ? e : e + 8) + 8 * (lane >> 4);
      b0[e] = Vs[kk * 32 + ko][lane & 15];
      b1[e] = Vs[kk * 32 + ko][16 + (lane & 15)];
    }
    o0 = wmma_bf16(ap, b0, o0);
    o1 = wmma_bf16(ap, b1, o1);
  }
#pragma unroll
  for (int r = 0; r < 8; ++r) {
    const int q = wid * 16 + r + 8 * (lane >> 4);
    const long long base = (((long long)(b * Nn + q)) * Tt + t) * Cc + h * 32;
    O[base + (lane & 15)]      = o0[r];
    O[base + 16 + (lane & 15)] = o1[r];
  }
}

// ---------------------------------------------------------------------------
// Elementwise kernels
// ---------------------------------------------------------------------------
__global__ __launch_bounds__(256) void add_pe_k(const float* __restrict__ q,
                                                const float* __restrict__ ds,
                                                float* __restrict__ out) {
  const long long i = (long long)blockIdx.x * 256 + threadIdx.x;
  if (i < SZ) {
    const int c = (int)(i & (Cc - 1));
    const int n = (int)((i >> 14) & (Nn - 1));   // i / (T*C) % N ; T*C = 2^14
    out[i] = q[i] + ds[n * Cc + c];
  }
}

// out = LN(A + R) * g + b over C=256; one wave per row, 8 rows per block
__global__ __launch_bounds__(256) void ln_res_k(const float* __restrict__ A,
                                                const float* __restrict__ R,
                                                const float* __restrict__ g,
                                                const float* __restrict__ bta,
                                                float* __restrict__ out) {
  const int lane = threadIdx.x & 31;
  const long long row = (long long)blockIdx.x * 8 + (threadIdx.x >> 5);
  const long long base = row * Cc;
  float x[8], sum = 0.f;
#pragma unroll
  for (int j = 0; j < 8; ++j) {
    const int c = lane + j * 32;
    x[j] = A[base + c] + R[base + c];
    sum += x[j];
  }
  const float mean = rsum32(sum) * (1.f / Cc);
  float var = 0.f;
#pragma unroll
  for (int j = 0; j < 8; ++j) { const float d = x[j] - mean; var += d * d; }
  var = rsum32(var) * (1.f / Cc);
  const float inv = rsqrtf(var + 1e-5f);
#pragma unroll
  for (int j = 0; j < 8; ++j) {
    const int c = lane + j * 32;
    out[base + c] = (x[j] - mean) * inv * g[c] + bta[c];
  }
}

__global__ __launch_bounds__(256) void gate_k(const float* __restrict__ fsU,
                                              const float* __restrict__ fgX,
                                              float* __restrict__ out) {
  const long long i = (long long)blockIdx.x * 256 + threadIdx.x;
  if (i < SZ) {
    const float a = fsU[i], x = fgX[i];
    const float gg = 1.f / (1.f + __expf(-(a + x)));
    out[i] = gg * a + (1.f - gg) * x;
  }
}

// ---------------------------------------------------------------------------
// Host launch
// ---------------------------------------------------------------------------
extern "C" void kernel_launch(void* const* d_in, const int* in_sizes, int n_in,
                              void* d_out, int out_size, void* d_ws,
                              size_t ws_size, hipStream_t stream) {
  const float* value = (const float*)d_in[0];
  const float* key_t = (const float*)d_in[1];
  const float* query = (const float*)d_in[2];
  const float* D_S   = (const float*)d_in[3];
  const float* gso   = (const float*)d_in[4];
  const float* We    = (const float*)d_in[5];
  const float* Wv    = (const float*)d_in[6];
  const float* Wk    = (const float*)d_in[7];
  const float* Wq    = (const float*)d_in[8];
  const float* Wo    = (const float*)d_in[9];
  const float* bo    = (const float*)d_in[10];
  const float* g1    = (const float*)d_in[11];
  const float* b1    = (const float*)d_in[12];
  const float* g2    = (const float*)d_in[13];
  const float* b2    = (const float*)d_in[14];
  const float* Wf1   = (const float*)d_in[15];
  const float* bf1   = (const float*)d_in[16];
  const float* Wf2   = (const float*)d_in[17];
  const float* bf2   = (const float*)d_in[18];
  const float* Wg    = (const float*)d_in[19];
  const float* bg    = (const float*)d_in[20];
  const float* Wfs   = (const float*)d_in[21];
  const float* bfs   = (const float*)d_in[22];
  const float* Wfg   = (const float*)d_in[23];
  const float* bfg   = (const float*)d_in[24];

  char* w = (char*)d_ws;
  auto alloc = [&](size_t bytes) -> void* {
    void* p = (void*)w;
    w += (bytes + 255) & ~(size_t)255;
    return p;
  };
  float*  dsP  = (float*)alloc((size_t)Nn * Cc * 4);
  float*  qbuf = (float*)alloc((size_t)SZ * 4);
  float*  big1 = (float*)alloc((size_t)SZ * 4);
  float*  XG   = (float*)alloc((size_t)SZ * 4);
  __bf16* vp   = (__bf16*)alloc((size_t)SZ * 2);
  __bf16* kp   = (__bf16*)alloc((size_t)SZ * 2);
  __bf16* qp   = (__bf16*)alloc((size_t)SZ * 2);
  float*  big2 = (float*)alloc((size_t)SZ * 4);
  float*  xln  = (float*)alloc((size_t)SZ * 4);
  __bf16* h1b  = (__bf16*)alloc((size_t)SZ * FEe * 2);
  float*  fsU  = (float*)alloc((size_t)SZ * 4);
  float*  fgX  = (float*)alloc((size_t)SZ * 4);

  const int rows = Bb * Nn * Tt;                      // 65536
  const long long bstride = (long long)Nn * Tt * Cc;  // per-batch elements

  // 1) ds = D_S @ We        [128,256], K=128
  gemm_wmma_k<0, false, false><<<dim3(Cc / 128, 1, 1), 256, 0, stream>>>(
      D_S, We, dsP, nullptr, Nn, Cc, Nn, Nn, Cc, Cc, 0, 0, 0, 0);
  // 2) qbuf = query + ds
  add_pe_k<<<(unsigned)(SZ / 256), 256, 0, stream>>>(query, dsP, qbuf);
  // 3) big1 = query @ Wg    [65536,256], K=256
  gemm_wmma_k<0, false, false><<<dim3(Cc / 128, rows / 128, 1), 256, 0,
                                 stream>>>(
      query, Wg, big1, nullptr, rows, Cc, Cc, Cc, Cc, Cc, 0, 0, 0, 0);
  // 4) XG[b] = gso @ big1[b] + bg  (M=128, N=T*C=16384, K=128, batch=B)
  gemm_wmma_k<1, false, false><<<dim3((Tt * Cc) / 128, 1, Bb), 256, 0,
                                 stream>>>(
      gso, big1, XG, bg, Nn, Tt * Cc, Nn, Nn, Tt * Cc, Tt * Cc,
      0, bstride, bstride, Cc - 1);
  // 5) per-head projections -> bf16
  proj_heads_k<<<rows / 64, 256, 0, stream>>>(value, Wv, vp);
  proj_heads_k<<<rows / 64, 256, 0, stream>>>(key_t, Wk, kp);
  proj_heads_k<<<rows / 64, 256, 0, stream>>>(qbuf,  Wq, qp);
  // 6) attention -> big1 (reuse)
  attn_k<<<dim3(Hh, Tt, Bb), 256, 0, stream>>>(qp, kp, vp, big1);
  // 7) big2 = attn_out @ Wo + bo
  gemm_wmma_k<1, false, false><<<dim3(Cc / 128, rows / 128, 1), 256, 0,
                                 stream>>>(
      big1, Wo, big2, bo, rows, Cc, Cc, Cc, Cc, Cc, 0, 0, 0, Cc - 1);
  // 8) xln = LN(big2 + qbuf)
  ln_res_k<<<rows / 8, 256, 0, stream>>>(big2, qbuf, g1, b1, xln);
  // 9) h1b = relu(xln @ Wf1 + bf1)  -> bf16  [65536,1024], K=256
  gemm_wmma_k<2, false, true><<<dim3((FEe * Cc) / 128, rows / 128, 1), 256, 0,
                                stream>>>(
      xln, Wf1, h1b, bf1, rows, FEe * Cc, Cc, Cc, FEe * Cc, FEe * Cc,
      0, 0, 0, FEe * Cc - 1);
  // 10) big2 = h1b @ Wf2 + bf2      (A is bf16 -> async LDS staging, K=1024)
  gemm_wmma_k<1, true, false><<<dim3(Cc / 128, rows / 128, 1), 256, 0,
                                stream>>>(
      h1b, Wf2, big2, bf2, rows, Cc, FEe * Cc, FEe * Cc, Cc, Cc,
      0, 0, 0, Cc - 1);
  // 11) U = LN(big2 + xln) -> qbuf (reuse)
  ln_res_k<<<rows / 8, 256, 0, stream>>>(big2, xln, g2, b2, qbuf);
  // 12) fsU = U @ Wfs + bfs
  gemm_wmma_k<1, false, false><<<dim3(Cc / 128, rows / 128, 1), 256, 0,
                                 stream>>>(
      qbuf, Wfs, fsU, bfs, rows, Cc, Cc, Cc, Cc, Cc, 0, 0, 0, Cc - 1);
  // 13) fgX = XG @ Wfg + bfg
  gemm_wmma_k<1, false, false><<<dim3(Cc / 128, rows / 128, 1), 256, 0,
                                 stream>>>(
      XG, Wfg, fgX, bfg, rows, Cc, Cc, Cc, Cc, Cc, 0, 0, 0, Cc - 1);
  // 14) gated output
  gate_k<<<(unsigned)(SZ / 256), 256, 0, stream>>>(fsU, fgX, (float*)d_out);

  (void)in_sizes; (void)n_in; (void)out_size; (void)ws_size;
}
